// KGCN_4827543240786
// MI455X (gfx1250) — matrix operand order repaired and checked by reading, not compiled, not run on hardware
//
#include <hip/hip_runtime.h>
#include <hip/hip_bf16.h>

typedef __bf16 v16bf __attribute__((ext_vector_type(16)));
typedef float  v8f   __attribute__((ext_vector_type(8)));

#define TB 16          // batch elements per block
#define XLD 72         // padded LDS row stride (bf16 elems) to break bank conflicts
#define NUM_ENT_C 100000
#define ITEM_IN_KG 20000

// A fragment: 16x32 bf16, row-major source. lane 0-15: M=lane, K={0..7,16..23};
// lane 16-31: M=lane-16, K={8..15,24..31} (per CDNA5 ISA 16-bit A layout).
__device__ __forceinline__ v16bf load_a_frag(const __bf16* base, int lane, int k0) {
  const int row = lane & 15;
  const int kb  = (lane >> 4) * 8;
  const __bf16* pr = base + row * XLD + k0 + kb;
  v16bf a;
#pragma unroll
  for (int i = 0; i < 8; ++i) { a[i] = pr[i]; a[8 + i] = pr[16 + i]; }
  return a;
}

// B fragment: 32x16 bf16. lane holds N=lane%16; lane 0-15: K={0..7,16..23},
// lane 16-31: K={8..15,24..31} (matches A); W^T staged row-major in LDS.
__device__ __forceinline__ v16bf load_b_frag(const __bf16* sWt, int lane, int k0, int nt) {
  const int k = k0 + ((lane >> 4) << 3);
  const __bf16* p = sWt + k * XLD + nt * 16 + (lane & 15);
  v16bf b;
#pragma unroll
  for (int i = 0; i < 8; ++i) { b[i] = p[i * XLD]; b[8 + i] = p[(16 + i) * XLD]; }
  return b;
}

__global__ __launch_bounds__(256)
void kgcn_fused_kernel(const int* __restrict__ u, const int* __restrict__ v,
                       const float* __restrict__ usr_emb, const float* __restrict__ item_emb,
                       const float* __restrict__ ent_emb, const float* __restrict__ rel_emb,
                       const float* __restrict__ W, const float* __restrict__ bvec,
                       const int* __restrict__ adj_ent, const int* __restrict__ adj_rel,
                       float* __restrict__ out) {
  __shared__ __bf16 sWt[64 * XLD];     // W^T, bf16
  __shared__ float  sBias[64];
  __shared__ float  sUe[TB * 64];
  __shared__ __bf16 sX[144 * XLD];     // X then (in-place) H
  __shared__ int    sEnts1[TB * 8];
  __shared__ int    sEnts2[TB * 64];
  __shared__ float  sWgt[TB * 64];     // hop-1 softmax weights
  __shared__ float  sS0[TB * 8];       // hop-0 raw scores
  __shared__ float  sW0[TB * 8];       // hop-0 softmax weights (reused in iter 1)
  __shared__ __bf16 sX2[TB * XLD];
  __shared__ float  sIV[TB * 64];      // item_vec

  const int t    = threadIdx.x;
  const int e    = t >> 4;             // batch element within block
  const int j    = t & 15;             // lane within 16-thread group
  const int lane = t & 31;
  const int wave = t >> 5;
  const int bidx = blockIdx.x * TB + e;

  // ---- Phase 0: stage W^T, bias, user embedding, hop-1 indices ----
  for (int i = t; i < 64 * 64; i += 256) {
    int r = i >> 6, c = i & 63;
    sWt[c * XLD + r] = (__bf16)W[i];    // sWt[k][n] = W[n][k]
  }
  if (t < 64) sBias[t] = bvec[t];

  const int u_i = u[bidx];
  const int v_i = v[bidx];
  const int vclip = (v_i >= ITEM_IN_KG) ? NUM_ENT_C : v_i;
  for (int d = j; d < 64; d += 16) sUe[e * 64 + d] = usr_emb[u_i * 64 + d];
  if (j < 8) sEnts1[e * 8 + j] = adj_ent[vclip * 8 + j];
  __syncthreads();

  // ---- Phase 1: attention scores + softmax ----
  if (j < 8) {                                  // hop-1: thread owns m = j
    const int m = j;
    const int e1 = sEnts1[e * 8 + m];
    const float* uep = &sUe[e * 64];
    float sc[8];
#pragma unroll
    for (int n = 0; n < 8; ++n) {
      const int r1 = adj_rel[e1 * 8 + n];
      sEnts2[(e * 8 + m) * 8 + n] = adj_ent[e1 * 8 + n];
      const float* re = rel_emb + r1 * 64;
      float s = 0.f;
      for (int d = 0; d < 64; ++d) s += uep[d] * re[d];
      sc[n] = s;
    }
    float mx = sc[0];
#pragma unroll
    for (int n = 1; n < 8; ++n) mx = fmaxf(mx, sc[n]);
    float ssum = 0.f;
#pragma unroll
    for (int n = 0; n < 8; ++n) { sc[n] = __expf(sc[n] - mx); ssum += sc[n]; }
    const float inv = 1.f / ssum;
#pragma unroll
    for (int n = 0; n < 8; ++n) sWgt[(e * 8 + m) * 8 + n] = sc[n] * inv;
  } else {                                      // hop-0 raw scores: thread owns n = j-8
    const int n = j - 8;
    const int r0 = adj_rel[vclip * 8 + n];
    const float* uep = &sUe[e * 64];
    const float* re  = rel_emb + r0 * 64;
    float s = 0.f;
    for (int d = 0; d < 64; ++d) s += uep[d] * re[d];
    sS0[e * 8 + n] = s;
  }
  __syncthreads();
  if (j < 8) {                                  // hop-0 softmax (weights reused in iter 1)
    float mx = sS0[e * 8];
#pragma unroll
    for (int n = 1; n < 8; ++n) mx = fmaxf(mx, sS0[e * 8 + n]);
    float ssum = 0.f;
#pragma unroll
    for (int n = 0; n < 8; ++n) ssum += __expf(sS0[e * 8 + n] - mx);
    sW0[e * 8 + j] = __expf(sS0[e * 8 + j] - mx) / ssum;
  }
  __syncthreads();

  // ---- Phase 2: build X (144 x 64, bf16): rows e*8+m = hop-1 sv+agg, rows 128+e = hop-0 ----
#pragma unroll
  for (int dd = 0; dd < 4; ++dd) {
    const int d = j + dd * 16;
    for (int m = 0; m < 8; ++m) {
      float agg = 0.f;
#pragma unroll
      for (int n = 0; n < 8; ++n)
        agg += sWgt[(e * 8 + m) * 8 + n] * ent_emb[sEnts2[(e * 8 + m) * 8 + n] * 64 + d];
      const float sv = ent_emb[sEnts1[e * 8 + m] * 64 + d];
      sX[(e * 8 + m) * XLD + d] = (__bf16)(sv + agg);
    }
    float agg0 = 0.f;
#pragma unroll
    for (int n = 0; n < 8; ++n)
      agg0 += sW0[e * 8 + n] * ent_emb[sEnts1[e * 8 + n] * 64 + d];
    sX[(128 + e) * XLD + d] = (__bf16)(item_emb[v_i * 64 + d] + agg0);
  }
  __syncthreads();

  // ---- Phase 3: H = sigmoid(X @ W^T + b), M=144 N=64 K=64, bf16 WMMA ----
  v8f acc[5];
#pragma unroll
  for (int i = 0; i < 5; ++i) {
    const int tile = wave + i * 8;
    if (tile < 36) {
      const int mt = tile >> 2, nt = tile & 3;
      v8f c = {};
      v16bf a0 = load_a_frag(sX + mt * 16 * XLD, lane, 0);
      v16bf b0 = load_b_frag(sWt, lane, 0, nt);
      c = __builtin_amdgcn_wmma_f32_16x16x32_bf16(false, a0, false, b0, (short)0, c, false, false);
      v16bf a1 = load_a_frag(sX + mt * 16 * XLD, lane, 32);
      v16bf b1 = load_b_frag(sWt, lane, 32, nt);
      c = __builtin_amdgcn_wmma_f32_16x16x32_bf16(false, a1, false, b1, (short)0, c, false, false);
      acc[i] = c;
    }
  }
  __syncthreads();   // all reads of X complete; H overwrites X in place
#pragma unroll
  for (int i = 0; i < 5; ++i) {
    const int tile = wave + i * 8;
    if (tile < 36) {
      const int mt = tile >> 2, nt = tile & 3;
      const int mbase = mt * 16 + ((lane >> 4) << 3);
      const int col   = nt * 16 + (lane & 15);
#pragma unroll
      for (int r = 0; r < 8; ++r) {
        const float hv = 1.f / (1.f + __expf(-(acc[i][r] + sBias[col])));
        sX[(mbase + r) * XLD + col] = (__bf16)hv;
      }
    }
  }
  __syncthreads();

  // ---- Phase 4: x2 = h0 + sum_n w0[n] * h1[n] ----
#pragma unroll
  for (int dd = 0; dd < 4; ++dd) {
    const int d = j + dd * 16;
    float agg = 0.f;
#pragma unroll
    for (int n = 0; n < 8; ++n)
      agg += sW0[e * 8 + n] * (float)sX[(e * 8 + n) * XLD + d];
    sX2[e * XLD + d] = (__bf16)((float)sX[(128 + e) * XLD + d] + agg);
  }
  __syncthreads();

  // ---- Phase 5: item_vec = tanh(X2 @ W^T + b), M=16, 4 N-tiles on waves 0..3 ----
  if (wave < 4) {
    const int nt = wave;
    v8f c = {};
    v16bf a0 = load_a_frag(sX2, lane, 0);
    v16bf b0 = load_b_frag(sWt, lane, 0, nt);
    c = __builtin_amdgcn_wmma_f32_16x16x32_bf16(false, a0, false, b0, (short)0, c, false, false);
    v16bf a1 = load_a_frag(sX2, lane, 32);
    v16bf b1 = load_b_frag(sWt, lane, 32, nt);
    c = __builtin_amdgcn_wmma_f32_16x16x32_bf16(false, a1, false, b1, (short)0, c, false, false);
    const int mbase = (lane >> 4) << 3;
    const int col   = nt * 16 + (lane & 15);
#pragma unroll
    for (int r = 0; r < 8; ++r)
      sIV[(mbase + r) * 64 + col] = tanhf(c[r] + sBias[col]);
  }
  __syncthreads();

  // ---- Phase 6: out = sigmoid(dot(ue, item_vec)) ----
  float p = 0.f;
#pragma unroll
  for (int dd = 0; dd < 4; ++dd) {
    const int d = j + dd * 16;
    p += sUe[e * 64 + d] * sIV[e * 64 + d];
  }
  p += __shfl_xor(p, 8, 32);
  p += __shfl_xor(p, 4, 32);
  p += __shfl_xor(p, 2, 32);
  p += __shfl_xor(p, 1, 32);
  if (j == 0) out[bidx] = 1.f / (1.f + __expf(-p));
}

extern "C" void kernel_launch(void* const* d_in, const int* in_sizes, int n_in,
                              void* d_out, int out_size, void* d_ws, size_t ws_size,
                              hipStream_t stream) {
  (void)n_in; (void)d_ws; (void)ws_size; (void)out_size;
  const int*   u        = (const int*)d_in[0];
  const int*   v        = (const int*)d_in[1];
  const float* usr_emb  = (const float*)d_in[2];
  const float* item_emb = (const float*)d_in[3];
  const float* ent_emb  = (const float*)d_in[4];
  const float* rel_emb  = (const float*)d_in[5];
  const float* W        = (const float*)d_in[6];
  const float* b        = (const float*)d_in[7];
  const int*   adj_ent  = (const int*)d_in[8];
  const int*   adj_rel  = (const int*)d_in[9];
  float* out = (float*)d_out;

  const int B = in_sizes[0];
  const int blocks = (B + TB - 1) / TB;
  kgcn_fused_kernel<<<blocks, 256, 0, stream>>>(u, v, usr_emb, item_emb, ent_emb,
                                                rel_emb, W, b, adj_ent, adj_rel, out);
}
